// PTMLoss_82729660056292
// MI455X (gfx1250) — matrix operand order repaired and checked by reading, not compiled, not run on hardware
//
#include <hip/hip_runtime.h>

// Problem constants (match reference)
#define NROWS 8192
#define DDIM  128
#define TDIM  256                          // 32-wide tiles per side (8192/32)
#define NTASK ((TDIM * (TDIM + 1)) / 2)    // 32896 upper-tri (incl. diag) 32x32 tiles
#define WPB   8                            // waves per block (256 threads, wave32)
#define PAIR_BLOCKS ((NTASK + WPB - 1) / WPB)

typedef __bf16 bf16_t;
typedef bf16_t v16bf __attribute__((ext_vector_type(16)));
typedef float  v8f   __attribute__((ext_vector_type(8)));

union Frag { v16bf v; uint4 u[2]; };

__device__ __forceinline__ unsigned short f2bf_rn(float f) {
    unsigned int u = __float_as_uint(f);
    u += 0x7FFFu + ((u >> 16) & 1u);       // round-to-nearest-even to bf16
    return (unsigned short)(u >> 16);
}
__device__ __forceinline__ float bf2f(unsigned short h) {
    return __uint_as_float(((unsigned int)h) << 16);
}

__device__ __forceinline__ float pe_clamped(float inv) {
    // pe(d) = 1/d^3 - 1/d^2 = inv^2 * (inv - 1), clipped to [-1.5, 1000]
    float pe = inv * inv * (inv - 1.0f);
    return fminf(fmaxf(pe, -1.5f), 1000.0f);
}

__device__ __forceinline__ v8f wmma_bf16(v16bf a, v16bf b, v8f c) {
    return __builtin_amdgcn_wmma_f32_16x16x32_bf16(
        /*neg_a=*/false, a, /*neg_b=*/false, b,
        /*c_mod=*/(short)0, c, /*reuse_a=*/false, /*reuse_b=*/false);
}

// A-matrix 16x32 bf16 fragment (ISA 7.12.2): lane holds row M=(lane&15);
// lanes 0-15: K = [kb..kb+7] in v0-3, [kb+16..kb+23] in v4-7 with kb = (lane>>4)*8
__device__ __forceinline__ void load_a(Frag& f, const unsigned short* rowp, int lane) {
    const int kb = (lane >> 4) << 3;
    f.u[0] = *reinterpret_cast<const uint4*>(rowp + kb);
    f.u[1] = *reinterpret_cast<const uint4*>(rowp + kb + 16);
}
// B-matrix 32x16 bf16 fragment: lane holds col N=(lane&15);
// K = [kh..kh+7] in v0-3, [kh+16..kh+23] in v4-7 with kh = (lane>>4)*8
__device__ __forceinline__ void load_b(Frag& f, const unsigned short* rowp, int lane) {
    const int kh = (lane >> 4) << 3;
    f.u[0] = *reinterpret_cast<const uint4*>(rowp + kh);
    f.u[1] = *reinterpret_cast<const uint4*>(rowp + kh + 16);
}

__device__ __forceinline__ int tri_offset(int ti) {
    return ti * TDIM - ((ti * (ti - 1)) >> 1);
}

// ---------------------------------------------------------------------------
// Kernel 1: per-row prep. sq[i], loss0 term, bf16 hi/lo split of x.
// One 128-thread block per row.
// ---------------------------------------------------------------------------
__global__ __launch_bounds__(DDIM) void prep_kernel(
    const float* __restrict__ x,
    unsigned short* __restrict__ xh,
    unsigned short* __restrict__ xl,
    float* __restrict__ sq,
    float* __restrict__ loss0) {
    __shared__ float r2[DDIM];
    __shared__ float r2e[DDIM];
    const int row = blockIdx.x;
    const int t   = threadIdx.x;

    const float f  = x[(size_t)row * DDIM + t];
    const unsigned short h = f2bf_rn(f);
    const float fh = bf2f(h);
    const unsigned short l = f2bf_rn(f - fh);
    xh[(size_t)row * DDIM + t] = h;
    xl[(size_t)row * DDIM + t] = l;

    const float fe = f + 1e-6f;            // torch pairwise_distance eps
    r2[t]  = f * f;
    r2e[t] = fe * fe;
    __syncthreads();
    for (int s = DDIM / 2; s > 0; s >>= 1) {
        if (t < s) { r2[t] += r2[t + s]; r2e[t] += r2e[t + s]; }
        __syncthreads();
    }
    if (t == 0) {
        sq[row] = r2[0];
        const float inv = rsqrtf(r2e[0]);  // 1/d0
        loss0[row] = pe_clamped(inv);
    }
}

// ---------------------------------------------------------------------------
// Kernel 2: pairwise tiles. One wave per 32x32 upper-tri tile.
// Gram dot products via split-bf16 WMMA (hi*hi + hi*lo + lo*hi), f32 accum.
// ---------------------------------------------------------------------------
__global__ __launch_bounds__(256) void pair_kernel(
    const unsigned short* __restrict__ xh,
    const unsigned short* __restrict__ xl,
    const float* __restrict__ sq,
    float* __restrict__ partial) {
    __shared__ float red[256];
    const int tid  = threadIdx.x;
    const int lane = tid & 31;
    const int warp = tid >> 5;
    const int task = blockIdx.x * WPB + warp;   // wave-uniform

    float lsum = 0.0f;
    if (task < NTASK) {                          // wave-uniform branch: EXEC stays full
        // Decode upper-triangular tile coords (ti <= tj)
        const float fB = (float)(2 * TDIM + 1);
        float disc = fB * fB - 8.0f * (float)task;
        int ti = (int)floorf((fB - sqrtf(fmaxf(disc, 0.0f))) * 0.5f);
        ti = ti < 0 ? 0 : (ti > TDIM - 1 ? TDIM - 1 : ti);
        while (ti > 0 && tri_offset(ti) > task) --ti;
        while (ti < TDIM - 1 && tri_offset(ti + 1) <= task) ++ti;
        const int tj = ti + (task - tri_offset(ti));

        const int i0 = ti * 32, j0 = tj * 32;
        const int rsel = lane & 15;              // row/col selector within 16-tile

        v8f acc[2][2] = {};
        #pragma unroll
        for (int kc = 0; kc < DDIM; kc += 32) {
            Frag ah[2], al[2], bh[2], bl[2];
            #pragma unroll
            for (int t = 0; t < 2; ++t) {
                const size_t ra = (size_t)(i0 + t * 16 + rsel) * DDIM + kc;
                const size_t rb = (size_t)(j0 + t * 16 + rsel) * DDIM + kc;
                load_a(ah[t], xh + ra, lane);
                load_a(al[t], xl + ra, lane);
                load_b(bh[t], xh + rb, lane);
                load_b(bl[t], xl + rb, lane);
            }
            #pragma unroll
            for (int mi = 0; mi < 2; ++mi)
                #pragma unroll
                for (int ni = 0; ni < 2; ++ni) {
                    acc[mi][ni] = wmma_bf16(ah[mi].v, bh[ni].v, acc[mi][ni]);
                    acc[mi][ni] = wmma_bf16(ah[mi].v, bl[ni].v, acc[mi][ni]);
                    acc[mi][ni] = wmma_bf16(al[mi].v, bh[ni].v, acc[mi][ni]);
                }
        }

        // Epilogue: C/D layout — lane l, VGPR v: M = v + 8*(l>>4), N = l&15
        const int iofs = (lane >> 4) * 8;
        const int jcol = lane & 15;
        #pragma unroll
        for (int mi = 0; mi < 2; ++mi)
            #pragma unroll
            for (int ni = 0; ni < 2; ++ni)
                #pragma unroll
                for (int v = 0; v < 8; ++v) {
                    const int i = i0 + mi * 16 + iofs + v;
                    const int j = j0 + ni * 16 + jcol;
                    if (i < j) {
                        const float dot = acc[mi][ni][v];
                        float d2 = fmaxf(sq[i] + sq[j] - 2.0f * dot, 0.0f);
                        lsum += pe_clamped(rsqrtf(d2));
                    }
                }
    }

    red[tid] = lsum;
    __syncthreads();
    for (int s = 128; s > 0; s >>= 1) {
        if (tid < s) red[tid] += red[tid + s];
        __syncthreads();
    }
    if (tid == 0) partial[blockIdx.x] = red[0];
}

// ---------------------------------------------------------------------------
// Kernel 3: deterministic final reduction. Single 256-thread block.
// ---------------------------------------------------------------------------
__global__ __launch_bounds__(256) void final_kernel(
    const float* __restrict__ loss0,
    const float* __restrict__ partial,
    int nPartial,
    float* __restrict__ out) {
    __shared__ float red[256];
    const int t = threadIdx.x;
    float s0 = 0.0f, sp = 0.0f;
    for (int i = t; i < NROWS; i += 256) s0 += loss0[i];
    for (int i = t; i < nPartial; i += 256) sp += partial[i];
    const float invN = 1.0f / (float)NROWS;
    const float invC = 1.0f / (float)((long long)NROWS * (NROWS - 1) / 2);
    red[t] = s0 * invN + sp * invC;
    __syncthreads();
    for (int s = 128; s > 0; s >>= 1) {
        if (t < s) red[t] += red[t + s];
        __syncthreads();
    }
    if (t == 0) out[0] = red[0];
}

extern "C" void kernel_launch(void* const* d_in, const int* in_sizes, int n_in,
                              void* d_out, int out_size, void* d_ws, size_t ws_size,
                              hipStream_t stream) {
    const float* x = (const float*)d_in[0];
    float* out = (float*)d_out;

    // Workspace layout (~4.3 MB total)
    char* ws = (char*)d_ws;
    unsigned short* xh = (unsigned short*)ws;                               // 2 MB
    unsigned short* xl = (unsigned short*)(ws + (size_t)NROWS * DDIM * 2);  // 2 MB
    float* sq   = (float*)(ws + (size_t)NROWS * DDIM * 4);                  // 32 KB
    float* l0   = sq + NROWS;                                               // 32 KB
    float* part = l0 + NROWS;                                               // ~16 KB

    prep_kernel<<<NROWS, DDIM, 0, stream>>>(x, xh, xl, sq, l0);
    pair_kernel<<<PAIR_BLOCKS, 256, 0, stream>>>(xh, xl, sq, part);
    final_kernel<<<1, 256, 0, stream>>>(l0, part, PAIR_BLOCKS, out);
}
